// MambaG2G_27479200760106
// MI455X (gfx1250) — matrix operand order repaired and checked by reading, not compiled, not run on hardware
//
#include <hip/hip_runtime.h>
#include <hip/hip_bf16.h>
#include <math.h>

#define D_MODEL 96
#define D_INNER 192
#define D_STATE 8
#define DT_RANK 6
#define NDBC    (DT_RANK + 2 * D_STATE)   // 22
#define NB      8                          // batch
#define NCHUNK  64                         // scan chunks over L

typedef __attribute__((ext_vector_type(16))) _Float16 v16h;
typedef __attribute__((ext_vector_type(8)))  _Float16 v8h;
typedef __attribute__((ext_vector_type(8)))  float    v8f;

__device__ __forceinline__ float fast_sigmoid(float x) { return 1.f / (1.f + __expf(-x)); }

// ---------------------------------------------------------------------------
// K1: RMSNorm + xz = xn @ W_in^T   (M x 96) @ (96 x 384) via WMMA f16->f32
// Wave computes one 16x16 tile; K=96 -> 3 wmma steps (fully unrolled).
// ---------------------------------------------------------------------------
__global__ void __launch_bounds__(128)
k_rmsnorm_gemm_in(const float* __restrict__ u, const float* __restrict__ norm_w,
                  const float* __restrict__ W_in,
                  float* __restrict__ xraw, float* __restrict__ zbuf)
{
  const int lane = threadIdx.x;
  const int row  = lane & 15;
  const int hi   = lane >> 4;                       // half-wave selector
  const int nt   = blockIdx.y * 4 + threadIdx.y;    // 0..23 N-tile (wave-uniform)
  const size_t t = (size_t)blockIdx.x * 16 + row;   // token row for A

  // RMSNorm scale for this row: lanes (r) and (r+16) each sum 48 squares
  float ss = 0.f;
  const float* up = u + t * D_MODEL + hi * 48;
#pragma unroll
  for (int k = 0; k < 48; ++k) { float v = up[k]; ss += v * v; }
  ss += __shfl_xor(ss, 16, 32);
  const float scale = rsqrtf(ss * (1.f / D_MODEL) + 1e-5f);

  v8f acc = {};
  const float* ur = u + t * D_MODEL;
  const int ncol  = nt * 16 + row;                  // column for B fragment
#pragma unroll
  for (int kt = 0; kt < D_MODEL; kt += 32) {
    v16h a, b;
    // A 16x32 f16 layout: lane<16 holds K {0..7,16..23}, lane>=16 holds {8..15,24..31}
    const int ka = kt + hi * 8;
#pragma unroll
    for (int i = 0; i < 16; ++i) {
      const int k = ka + (i < 8 ? i : i + 8);
      a[i] = (_Float16)(ur[k] * scale * norm_w[k]);
    }
    // B 32x16: N = lane&15, K = kt + hi*8 + (i<8 ? i : i+8)
    const float* wr = W_in + (size_t)ncol * D_MODEL + kt + hi * 8;
#pragma unroll
    for (int i = 0; i < 16; ++i) b[i] = (_Float16)wr[i < 8 ? i : i + 8];
    acc = __builtin_amdgcn_wmma_f32_16x16x32_f16(false, a, false, b, (short)0, acc,
                                                 false, false);
  }
  // C/D layout: VGPR r -> M = 8*hi + r, N = lane&15.
  // x/z split is 16-aligned -> wave-uniform on nt: no per-lane EXEC games.
  float* __restrict__ dst = (nt < 12) ? xraw : zbuf;
  const int col = (nt < 12) ? ncol : (ncol - D_INNER);
  const size_t mbase = (size_t)blockIdx.x * 16 + hi * 8;
#pragma unroll
  for (int r = 0; r < 8; ++r) dst[(mbase + r) * D_INNER + col] = acc[r];
}

// ---------------------------------------------------------------------------
// K2: depthwise causal conv (K=3) + SiLU
// ---------------------------------------------------------------------------
__global__ void __launch_bounds__(256)
k_conv_silu(const float* __restrict__ xraw, const float* __restrict__ conv_w,
            const float* __restrict__ conv_b, float* __restrict__ xc,
            int L, size_t total)
{
  const size_t idx = (size_t)blockIdx.x * 256 + threadIdx.x;
  if (idx >= total) return;
  const int d    = (int)(idx % D_INNER);
  const size_t t = idx / D_INNER;
  const int l    = (int)(t % L);
  float acc = conv_b[d];
  acc += conv_w[d * 3 + 2] * xraw[t * D_INNER + d];
  if (l >= 1) acc += conv_w[d * 3 + 1] * xraw[(t - 1) * D_INNER + d];
  if (l >= 2) acc += conv_w[d * 3 + 0] * xraw[(t - 2) * D_INNER + d];
  xc[idx] = acc * fast_sigmoid(acc);
}

// ---------------------------------------------------------------------------
// K3: dbc = xc @ W_x^T (22 outputs, K=192, wave-butterfly reduce) then
//     delta = softplus(dt @ W_dt^T + b_dt); store Bm, Cm.  One wave per token.
// ---------------------------------------------------------------------------
__global__ void __launch_bounds__(256)
k_dbc_delta(const float* __restrict__ xc, const float* __restrict__ W_x,
            const float* __restrict__ W_dt, const float* __restrict__ b_dt,
            float* __restrict__ delta, float* __restrict__ Bm, float* __restrict__ Cm,
            size_t M)
{
  const int lane = threadIdx.x & 31;
  const size_t t = (size_t)blockIdx.x * 8 + (threadIdx.x >> 5);
  if (t >= M) return;
  float acc[NDBC];
#pragma unroll
  for (int c = 0; c < NDBC; ++c) acc[c] = 0.f;
  const float* xr = xc + t * D_INNER;
  for (int k = lane; k < D_INNER; k += 32) {
    const float xv = xr[k];
#pragma unroll
    for (int c = 0; c < NDBC; ++c) acc[c] += xv * W_x[c * D_INNER + k];
  }
#pragma unroll
  for (int off = 16; off > 0; off >>= 1) {
#pragma unroll
    for (int c = 0; c < NDBC; ++c) acc[c] += __shfl_xor(acc[c], off, 32);
  }
  // each lane produces 6 delta channels
#pragma unroll
  for (int j = 0; j < 6; ++j) {
    const int d = lane * 6 + j;
    float s = b_dt[d];
#pragma unroll
    for (int r = 0; r < DT_RANK; ++r) s += acc[r] * W_dt[d * DT_RANK + r];
    delta[t * D_INNER + d] = (s > 20.f) ? s : log1pf(__expf(s));
  }
  if (lane < D_STATE)          Bm[t * D_STATE + lane]       = acc[DT_RANK + lane];
  else if (lane < 2 * D_STATE) Cm[t * D_STATE + (lane - 8)] = acc[DT_RANK + lane];
}

// ---------------------------------------------------------------------------
// K4a: chunked scan pass 1 — local scan from h=0; emit chunk product P and h_end.
// ---------------------------------------------------------------------------
__global__ void __launch_bounds__(32)
k_scan_pass1(const float* __restrict__ delta, const float* __restrict__ xc,
             const float* __restrict__ Bm, const float* __restrict__ A_log,
             float* __restrict__ Pbuf, float* __restrict__ Hend, int L, int CL)
{
  const int lane = threadIdx.x;
  const int d = blockIdx.x * 32 + lane;
  const int c = blockIdx.y;
  const int b = blockIdx.z;
  float a[D_STATE];
#pragma unroll
  for (int s = 0; s < D_STATE; ++s) a[s] = -__expf(A_log[d * D_STATE + s]);
  float h[D_STATE], p[D_STATE];
#pragma unroll
  for (int s = 0; s < D_STATE; ++s) { h[s] = 0.f; p[s] = 1.f; }
  const size_t tbase = (size_t)b * L + (size_t)c * CL;
  for (int i = 0; i < CL; ++i) {
    const size_t t = tbase + i;
    if (i + 1 < CL) {
      __builtin_prefetch(delta + (t + 1) * D_INNER + d, 0, 1);
      __builtin_prefetch(xc    + (t + 1) * D_INNER + d, 0, 1);
    }
    const float dl = delta[t * D_INNER + d];
    const float dx = dl * xc[t * D_INNER + d];
    const float* Bp = Bm + t * D_STATE;
#pragma unroll
    for (int s = 0; s < D_STATE; ++s) {
      const float dA = __expf(dl * a[s]);
      h[s] = dA * h[s] + dx * Bp[s];
      p[s] *= dA;
    }
  }
  const size_t off = ((((size_t)b * D_INNER) + d) * NCHUNK + c) * D_STATE;
#pragma unroll
  for (int s = 0; s < D_STATE; ++s) { Pbuf[off + s] = p[s]; Hend[off + s] = h[s]; }
}

// ---------------------------------------------------------------------------
// K4b: sequential combine across NCHUNK chunks (tiny): h_start per chunk.
// ---------------------------------------------------------------------------
__global__ void __launch_bounds__(32)
k_scan_combine(const float* __restrict__ Pbuf, const float* __restrict__ Hend,
               float* __restrict__ Hstart)
{
  const int d = blockIdx.x * 32 + threadIdx.x;
  const int b = blockIdx.y;
  float h[D_STATE];
#pragma unroll
  for (int s = 0; s < D_STATE; ++s) h[s] = 0.f;
  for (int c = 0; c < NCHUNK; ++c) {
    const size_t off = ((((size_t)b * D_INNER) + d) * NCHUNK + c) * D_STATE;
#pragma unroll
    for (int s = 0; s < D_STATE; ++s) {
      Hstart[off + s] = h[s];
      h[s] = Pbuf[off + s] * h[s] + Hend[off + s];
    }
  }
}

// ---------------------------------------------------------------------------
// K4c: chunked scan pass 2 — re-scan seeded with h_start, produce
//      y_out = (sum_s h*C + D*x) * silu(z), stored f16 for the out GEMM.
// ---------------------------------------------------------------------------
__global__ void __launch_bounds__(32)
k_scan_pass2(const float* __restrict__ delta, const float* __restrict__ xc,
             const float* __restrict__ zbuf, const float* __restrict__ Bm,
             const float* __restrict__ Cm, const float* __restrict__ A_log,
             const float* __restrict__ D_ssm, const float* __restrict__ Hstart,
             _Float16* __restrict__ yh, int L, int CL)
{
  const int lane = threadIdx.x;
  const int d = blockIdx.x * 32 + lane;
  const int c = blockIdx.y;
  const int b = blockIdx.z;
  float a[D_STATE];
#pragma unroll
  for (int s = 0; s < D_STATE; ++s) a[s] = -__expf(A_log[d * D_STATE + s]);
  const float Dd = D_ssm[d];
  float h[D_STATE];
  const size_t off = ((((size_t)b * D_INNER) + d) * NCHUNK + c) * D_STATE;
#pragma unroll
  for (int s = 0; s < D_STATE; ++s) h[s] = Hstart[off + s];
  const size_t tbase = (size_t)b * L + (size_t)c * CL;
  for (int i = 0; i < CL; ++i) {
    const size_t t = tbase + i;
    if (i + 1 < CL) {
      __builtin_prefetch(delta + (t + 1) * D_INNER + d, 0, 1);
      __builtin_prefetch(xc    + (t + 1) * D_INNER + d, 0, 1);
    }
    const float dl = delta[t * D_INNER + d];
    const float xv = xc[t * D_INNER + d];
    const float zv = zbuf[t * D_INNER + d];
    const float dx = dl * xv;
    const float* Bp = Bm + t * D_STATE;
    const float* Cp = Cm + t * D_STATE;
    float y = Dd * xv;
#pragma unroll
    for (int s = 0; s < D_STATE; ++s) {
      const float dA = __expf(dl * a[s]);
      h[s] = dA * h[s] + dx * Bp[s];
      y += h[s] * Cp[s];
    }
    yh[t * D_INNER + d] = (_Float16)(y * zv * fast_sigmoid(zv));
  }
}

// ---------------------------------------------------------------------------
// K5: e = u + y @ W_out^T   (M x 192) @ (192 x 96) via WMMA, K=192 -> 6 steps
// A fragment loaded as two 16B vector loads per k-step (layout is contiguous
// per half-wave: K {ka..ka+7} and {ka+16..ka+23}).
// ---------------------------------------------------------------------------
__global__ void __launch_bounds__(192)
k_gemm_out(const _Float16* __restrict__ yh, const float* __restrict__ W_out,
           const float* __restrict__ u, float* __restrict__ e)
{
  const int lane = threadIdx.x;
  const int row  = lane & 15;
  const int hi   = lane >> 4;
  const int nt   = threadIdx.y;                     // 0..5
  const size_t t = (size_t)blockIdx.x * 16 + row;
  const int ncol = nt * 16 + row;
  v8f acc = {};
  const v8h* yr8 = (const v8h*)(yh + t * D_INNER);  // 16B-aligned 8-half chunks
#pragma unroll
  for (int kt = 0; kt < D_INNER; kt += 32) {
    const int cbase = (kt >> 3) + hi;               // chunk of K {ka..ka+7}
    const v8h alo = yr8[cbase];
    const v8h ahi = yr8[cbase + 2];                 // K {ka+16..ka+23}
    const v16h a = __builtin_shufflevector(alo, ahi, 0, 1, 2, 3, 4, 5, 6, 7,
                                           8, 9, 10, 11, 12, 13, 14, 15);
    v16h b;
    const float* wr = W_out + (size_t)ncol * D_INNER + kt + hi * 8;
#pragma unroll
    for (int i = 0; i < 16; ++i) b[i] = (_Float16)wr[i < 8 ? i : i + 8];
    acc = __builtin_amdgcn_wmma_f32_16x16x32_f16(false, a, false, b, (short)0, acc,
                                                 false, false);
  }
  const size_t mbase = (size_t)blockIdx.x * 16 + hi * 8;
#pragma unroll
  for (int r = 0; r < 8; ++r) {
    const size_t mrow = mbase + r;
    e[mrow * D_MODEL + ncol] = acc[r] + u[mrow * D_MODEL + ncol];
  }
}

// ---------------------------------------------------------------------------
// K6: deterministic mean over L: one block per (b, col)
// ---------------------------------------------------------------------------
__global__ void __launch_bounds__(256)
k_mean(const float* __restrict__ e, float* __restrict__ em, int L)
{
  const int b    = blockIdx.x / D_MODEL;
  const int ccol = blockIdx.x % D_MODEL;
  __shared__ float sm[256];
  float s = 0.f;
  for (int l = threadIdx.x; l < L; l += 256)
    s += e[((size_t)b * L + l) * D_MODEL + ccol];
  sm[threadIdx.x] = s;
  __syncthreads();
  for (int off = 128; off > 0; off >>= 1) {
    if (threadIdx.x < off) sm[threadIdx.x] += sm[threadIdx.x + off];
    __syncthreads();
  }
  if (threadIdx.x == 0) em[b * D_MODEL + ccol] = sm[0] / (float)L;
}

// ---------------------------------------------------------------------------
// K7: head — fc(96->256)+tanh+elu, then mu/sigma (256->64). One block per b.
// ---------------------------------------------------------------------------
__global__ void __launch_bounds__(256)
k_head(const float* __restrict__ em, const float* __restrict__ fc_w,
       const float* __restrict__ fc_b, const float* __restrict__ mu_w,
       const float* __restrict__ mu_b, const float* __restrict__ sg_w,
       const float* __restrict__ sg_b, float* __restrict__ out)
{
  const int b = blockIdx.x;
  const int j = threadIdx.x;
  __shared__ float hsh[256];
  float s = fc_b[j];
  const float* er = em + b * D_MODEL;
#pragma unroll 4
  for (int k = 0; k < D_MODEL; ++k) s += er[k] * fc_w[j * D_MODEL + k];
  const float th = tanhf(s);
  hsh[j] = th > 0.f ? th : (__expf(th) - 1.f);        // elu(tanh(.))
  __syncthreads();
  if (j < 64) {
    float m = mu_b[j];
    for (int k = 0; k < 256; ++k) m += hsh[k] * mu_w[j * 256 + k];
    out[b * 64 + j] = m;
  } else if (j < 128) {
    const int o = j - 64;
    float g = sg_b[o];
    for (int k = 0; k < 256; ++k) g += hsh[k] * sg_w[o * 256 + k];
    g = g > 0.f ? g : (__expf(g) - 1.f);
    out[NB * 64 + b * 64 + o] = g + 1.0f + 1e-14f;
  }
}

// ---------------------------------------------------------------------------
extern "C" void kernel_launch(void* const* d_in, const int* in_sizes, int n_in,
                              void* d_out, int out_size, void* d_ws, size_t ws_size,
                              hipStream_t stream)
{
  const float* u      = (const float*)d_in[0];
  const float* norm_w = (const float*)d_in[1];
  const float* W_in   = (const float*)d_in[2];
  const float* conv_w = (const float*)d_in[3];
  const float* conv_b = (const float*)d_in[4];
  const float* W_x    = (const float*)d_in[5];
  const float* W_dt   = (const float*)d_in[6];
  const float* b_dt   = (const float*)d_in[7];
  const float* A_log  = (const float*)d_in[8];
  const float* D_ssm  = (const float*)d_in[9];
  const float* W_out  = (const float*)d_in[10];
  const float* fc_w   = (const float*)d_in[11];
  const float* fc_b   = (const float*)d_in[12];
  const float* mu_w   = (const float*)d_in[13];
  const float* mu_b   = (const float*)d_in[14];
  const float* sg_w   = (const float*)d_in[15];
  const float* sg_b   = (const float*)d_in[16];
  float* out = (float*)d_out;

  const size_t M = (size_t)in_sizes[0] / D_MODEL;   // B*L tokens (32768)
  const int L  = (int)(M / NB);                     // 4096
  const int CL = L / NCHUNK;                        // 64

  // workspace carve-out (256B aligned)
  char* w = (char*)d_ws;
  size_t o = 0;
  auto carve = [&](size_t bytes) -> char* {
    char* p = w + o;
    o += (bytes + 255) & ~(size_t)255;
    return p;
  };
  float*    xraw = (float*)carve(M * D_INNER * 4);
  float*    zbuf = (float*)carve(M * D_INNER * 4);
  float*    xc   = (float*)carve(M * D_INNER * 4);
  float*    Bm   = (float*)carve(M * D_STATE * 4);
  float*    Cm   = (float*)carve(M * D_STATE * 4);
  _Float16* yh   = (_Float16*)carve(M * D_INNER * 2);
  const size_t scanN = (size_t)NB * D_INNER * NCHUNK * D_STATE;
  float* Pbuf   = (float*)carve(scanN * 4);
  float* Hend   = (float*)carve(scanN * 4);
  float* Hstart = (float*)carve(scanN * 4);
  float* em     = (float*)carve((size_t)NB * D_MODEL * 4);
  float* delta  = xraw;   // xraw dead after conv -> reuse for delta
  float* e      = zbuf;   // z dead after scan pass 2 -> reuse for e

  // 1) RMSNorm + input GEMM (WMMA)
  k_rmsnorm_gemm_in<<<dim3((unsigned)(M / 16), 6), dim3(32, 4), 0, stream>>>(
      u, norm_w, W_in, xraw, zbuf);
  // 2) depthwise causal conv + SiLU
  const size_t total = M * D_INNER;
  k_conv_silu<<<(unsigned)((total + 255) / 256), 256, 0, stream>>>(
      xraw, conv_w, conv_b, xc, L, total);
  // 3) dbc + delta
  k_dbc_delta<<<(unsigned)((M + 7) / 8), 256, 0, stream>>>(
      xc, W_x, W_dt, b_dt, delta, Bm, Cm, M);
  // 4) chunked parallel scan (3 passes)
  k_scan_pass1<<<dim3(D_INNER / 32, NCHUNK, NB), 32, 0, stream>>>(
      delta, xc, Bm, A_log, Pbuf, Hend, L, CL);
  k_scan_combine<<<dim3(D_INNER / 32, NB), 32, 0, stream>>>(Pbuf, Hend, Hstart);
  k_scan_pass2<<<dim3(D_INNER / 32, NCHUNK, NB), 32, 0, stream>>>(
      delta, xc, zbuf, Bm, Cm, A_log, D_ssm, Hstart, yh, L, CL);
  // 5) output GEMM + residual (WMMA)
  k_gemm_out<<<(unsigned)(M / 16), dim3(32, 6), 0, stream>>>(yh, W_out, u, e);
  // 6) mean over L
  k_mean<<<NB * D_MODEL, 256, 0, stream>>>(e, em, L);
  // 7) head
  k_head<<<NB, 256, 0, stream>>>(em, fc_w, fc_b, mu_w, mu_b, sg_w, sg_b, out);

  (void)n_in; (void)out_size; (void)ws_size;
}
